// AttnBlock_79405355369062
// MI455X (gfx1250) — matrix operand order repaired and checked by reading, not compiled, not run on hardware
//
#include <hip/hip_runtime.h>
#include <hip/hip_bf16.h>
#include <cstdint>

// ---------------------------------------------------------------------------
// AttnBlock (GroupNorm + single-head attention over 4096 tokens, C=512) for
// gfx1250. All GEMM-like work goes through v_wmma_f32_16x16x32_f16; the flash
// attention K-chunks are staged to LDS with async global->LDS DMA (ASYNCcnt).
// ---------------------------------------------------------------------------

typedef __attribute__((ext_vector_type(16))) _Float16 v16h;
typedef __attribute__((ext_vector_type(8)))  _Float16 v8h;
typedef __attribute__((ext_vector_type(8)))  float    v8f;
typedef int v4i_ __attribute__((vector_size(16)));   // matches builtin param type

#define CC 512      // channels
#define NN 4096     // tokens (64*64)
#define BB 4        // batch

#define WMMA_F16(a, b, c) \
  __builtin_amdgcn_wmma_f32_16x16x32_f16(false, (a), false, (b), (short)0, (c), false, false)

// --- async global->LDS path (guarded; falls back to synchronous copy) ------
#if defined(__has_builtin)
#  if __has_builtin(__builtin_amdgcn_global_load_async_to_lds_b128)
#    define HAS_ASYNC_LDS 1
#  endif
#endif
#ifndef HAS_ASYNC_LDS
#  define HAS_ASYNC_LDS 0
#endif

__device__ __forceinline__ void copy16_g2l(const void* gsrc, void* ldst) {
#if HAS_ASYNC_LDS
  // Generic LDS address low 32 bits == LDS offset (ISA 10.2); generic global
  // address == AS(1) address. Reinterpret through uintptr_t.
  __builtin_amdgcn_global_load_async_to_lds_b128(
      (__attribute__((address_space(1))) v4i_*)(uintptr_t)gsrc,
      (__attribute__((address_space(3))) v4i_*)(uint32_t)(uintptr_t)ldst,
      0, 0);
#else
  *(v8h*)ldst = *(const v8h*)gsrc;
#endif
}

__device__ __forceinline__ void async_wait0() {
#if HAS_ASYNC_LDS
#  if __has_builtin(__builtin_amdgcn_s_wait_asynccnt)
  __builtin_amdgcn_s_wait_asynccnt(0);
#  else
  asm volatile("s_wait_asynccnt 0" ::: "memory");
#  endif
#endif
}

// A fragment: 16x32 f16 tile, rows of a row-major [*, ld] matrix.
// Contiguous k order (same as load_b_frag): lane<16 -> row=lane, K = {0..15}; lane>=16 -> K = {16..31}.
__device__ __forceinline__ v16h load_a_frag(const _Float16* base, int k0, int ld, int lane) {
  const int hl = lane >> 4;
  const int r  = lane & 15;
  const _Float16* p = base + r * ld + k0 + hl * 16;
  union { v16h v; struct { v8h lo, hi; } s; } u;
  u.s.lo = *(const v8h*)p;
  u.s.hi = *(const v8h*)(p + 8);
  return u.v;
}

// B fragment: 32x16 f16 tile; B(k, col) taken from memory laid out as
// mem[col][k] (K contiguous per column, column stride ld).
__device__ __forceinline__ v16h load_b_frag(const _Float16* base, int col0, int k0,
                                            int ld, int lane) {
  const int hl = lane >> 4;
  const int cn = lane & 15;
  const _Float16* p = base + (size_t)(col0 + cn) * ld + k0 + hl * 16;
  return *(const v16h*)p;
}

// ---------------------------------------------------------------------------
// GroupNorm pass 1: per (batch, group) mean / rstd. One block per (b,g);
// a group = 16 channels * 4096 pixels = 65536 contiguous floats.
// ---------------------------------------------------------------------------
__global__ __launch_bounds__(256) void gn_stats_kernel(const float* __restrict__ x,
                                                       float* __restrict__ stats) {
  __shared__ float rs[256];
  __shared__ float rq[256];
  const float4* p = (const float4*)(x + ((size_t)blockIdx.x << 16));
  float s = 0.f, q = 0.f;
  for (int i = threadIdx.x; i < 16384; i += 256) {
    float4 v = p[i];
    s += v.x + v.y + v.z + v.w;
    q += v.x * v.x + v.y * v.y + v.z * v.z + v.w * v.w;
  }
  rs[threadIdx.x] = s;
  rq[threadIdx.x] = q;
  __syncthreads();
  for (int st = 128; st > 0; st >>= 1) {
    if (threadIdx.x < st) {
      rs[threadIdx.x] += rs[threadIdx.x + st];
      rq[threadIdx.x] += rq[threadIdx.x + st];
    }
    __syncthreads();
  }
  if (threadIdx.x == 0) {
    float mean = rs[0] * (1.0f / 65536.0f);
    float var  = rq[0] * (1.0f / 65536.0f) - mean * mean;
    stats[blockIdx.x * 2 + 0] = mean;
    stats[blockIdx.x * 2 + 1] = rsqrtf(var + 1e-6f);
  }
}

// ---------------------------------------------------------------------------
// GroupNorm pass 2: normalize and transpose [B,C,N] -> token-major f16 [B,N,C].
// ---------------------------------------------------------------------------
__global__ __launch_bounds__(256) void gn_apply_kernel(const float* __restrict__ x,
                                                       const float* __restrict__ g,
                                                       const float* __restrict__ beta,
                                                       const float* __restrict__ stats,
                                                       _Float16* __restrict__ th) {
  int idx = blockIdx.x * 256 + threadIdx.x;           // 4*512*4096 total
  int n = idx & 4095;
  int c = (idx >> 12) & 511;
  int b = idx >> 21;
  int sg = (b << 5) | (c >> 4);
  float mean = stats[sg * 2 + 0];
  float rstd = stats[sg * 2 + 1];
  float v = (x[idx] - mean) * rstd * g[c] + beta[c];
  th[(((size_t)b << 12) + n) * CC + c] = (_Float16)v;
}

// ---------------------------------------------------------------------------
// Weight f32 -> f16 convert (wq|wk|wv|wo concatenated, layout preserved [o][c]).
// ---------------------------------------------------------------------------
__global__ __launch_bounds__(256) void wconvert_kernel(const float* __restrict__ wq,
                                                       const float* __restrict__ wk,
                                                       const float* __restrict__ wv,
                                                       const float* __restrict__ wo,
                                                       _Float16* __restrict__ wh) {
  int idx = blockIdx.x * 256 + threadIdx.x;           // 4 * 512*512 total
  const float* src = (idx < 262144) ? wq
                   : (idx < 524288) ? wk
                   : (idx < 786432) ? wv : wo;
  wh[idx] = (_Float16)src[idx & 262143];
}

// ---------------------------------------------------------------------------
// QKV GEMM: out[b,n,o] = sum_c t[b,n,c] * w[o,c] + bias[o].
// Grid: (colStrip=8, rowStrip=64, b*3+m). 128 threads; each wave: 16 rows x 64 cols.
// Q,K written token-major f16 [B,N,C]; V written transposed f16 [B,C,N].
// ---------------------------------------------------------------------------
__global__ __launch_bounds__(128) void gemm_qkv_kernel(const _Float16* __restrict__ th,
                                                       const _Float16* __restrict__ wh,
                                                       const float* __restrict__ bq,
                                                       const float* __restrict__ bk,
                                                       const float* __restrict__ bv,
                                                       _Float16* __restrict__ qh,
                                                       _Float16* __restrict__ kh,
                                                       _Float16* __restrict__ vT) {
  const int m = blockIdx.z % 3;
  const int b = blockIdx.z / 3;
  const int colBase = blockIdx.x * 64;
  const int wave = threadIdx.x >> 5;
  const int lane = threadIdx.x & 31;
  const int hl = lane >> 4, ln = lane & 15;
  const int row0 = blockIdx.y * 64 + wave * 16;

  const _Float16* A = th + ((size_t)b * NN + row0) * CC;
  const _Float16* W = wh + (size_t)m * CC * CC;
  const float* bias = (m == 0) ? bq : (m == 1) ? bk : bv;

  v8f zero = {};
  v8f acc[4] = {zero, zero, zero, zero};
#pragma unroll
  for (int kk = 0; kk < CC; kk += 32) {
    v16h a = load_a_frag(A, kk, CC, lane);
#pragma unroll
    for (int t = 0; t < 4; ++t) {
      v16h bf = load_b_frag(W, colBase + t * 16, kk, CC, lane);
      acc[t] = WMMA_F16(a, bf, acc[t]);
    }
  }

  if (m < 2) {
    _Float16* out = ((m == 0) ? qh : kh) + ((size_t)b * NN + row0) * CC;
#pragma unroll
    for (int t = 0; t < 4; ++t) {
      int col = colBase + t * 16 + ln;
      float bb = bias[col];
#pragma unroll
      for (int r = 0; r < 8; ++r)
        out[(size_t)(r + 8 * hl) * CC + col] = (_Float16)(acc[t][r] + bb);
    }
  } else {
    _Float16* out = vT + (size_t)b * CC * NN;
#pragma unroll
    for (int t = 0; t < 4; ++t) {
      int col = colBase + t * 16 + ln;
      float bb = bias[col];
#pragma unroll
      for (int r = 0; r < 8; ++r)
        out[(size_t)col * NN + row0 + r + 8 * hl] = (_Float16)(acc[t][r] + bb);
    }
  }
}

// ---------------------------------------------------------------------------
// Flash attention: one workgroup (4 waves) per 16 query rows per batch.
// K chunks (32 keys x 512 ch = 32 KB, contiguous) are staged to LDS with
// double-buffered async global->LDS DMA; the next chunk's transfer overlaps
// the current chunk's 40 WMMAs. All 4 waves compute the shared 16x32 score
// tile from LDS (softmax stats stay wave-local); each wave owns a disjoint
// 128-wide strip of the 512 output channels for the P*V accumulation.
// ---------------------------------------------------------------------------
__global__ __launch_bounds__(128) void flash_attn_kernel(const _Float16* __restrict__ qh,
                                                         const _Float16* __restrict__ kh,
                                                         const _Float16* __restrict__ vT,
                                                         _Float16* __restrict__ hh) {
  __shared__ alignas(16) _Float16 lds_q[16 * CC];       // 16 KB query tile
  __shared__ alignas(16) _Float16 lds_k[2][32 * CC];    // 2 x 32 KB K chunks
  __shared__ alignas(16) _Float16 lds_p[4][16 * 32];    // per-wave P staging

  const int b = blockIdx.y;
  const int row0 = blockIdx.x * 16;
  const int tid = threadIdx.x;
  const int wave = tid >> 5;
  const int lane = tid & 31;
  const int hl = lane >> 4, ln = lane & 15;
  const int colBase = wave * 128;

  const _Float16* kb = kh + (size_t)b * NN * CC;   // [n][c] : B operand for Q.K^T
  const _Float16* vb = vT + (size_t)b * CC * NN;   // [c][n] : B operand for P.V

  // Stage the 16x512 Q tile into LDS (cooperative, vectorized).
  {
    const v8h* src = (const v8h*)(qh + ((size_t)b * NN + row0) * CC);
    v8h* dst = (v8h*)lds_q;
    for (int i = tid; i < 16 * CC / 8; i += 128) dst[i] = src[i];
  }

  // Issue the first K-chunk DMA (32 KB linear copy, 16 b128 issues/thread).
  {
    const char* s = (const char*)kb;
    char* d = (char*)&lds_k[0][0];
#pragma unroll
    for (int i = 0; i < 16; ++i) {
      int off = (tid + i * 128) * 16;
      copy16_g2l(s + off, d + off);
    }
  }

  v8f zero = {};
  v8f o[8];
  float m[8], l[8];
#pragma unroll
  for (int t = 0; t < 8; ++t) o[t] = zero;
#pragma unroll
  for (int r = 0; r < 8; ++r) { m[r] = -1e30f; l[r] = 0.f; }

  const float scale = 0.044194173824159216f;  // 512^-0.5

  for (int it = 0; it < NN / 32; ++it) {
    const int j0 = it * 32;
    const int cur = it & 1;

    async_wait0();       // current K chunk resident in LDS
    __syncthreads();     // ... and visible to (done being written for) all waves

    // Prefetch the next K chunk into the other buffer while we compute.
    if (it + 1 < NN / 32) {
      const char* s = (const char*)(kb + (size_t)(j0 + 32) * CC);
      char* d = (char*)&lds_k[cur ^ 1][0];
#pragma unroll
      for (int i = 0; i < 16; ++i) {
        int off = (tid + i * 128) * 16;
        copy16_g2l(s + off, d + off);
      }
    }

    const _Float16* kt = &lds_k[cur][0];   // [32 keys][512 ch]

    // ---- S tile: 16 rows x 32 keys, contraction over C=512 --------------
    v8f s0 = zero, s1 = zero;
#pragma unroll
    for (int kk = 0; kk < CC; kk += 32) {
      v16h a  = load_a_frag(lds_q, kk, CC, lane);
      v16h b0 = load_b_frag(kt, 0,  kk, CC, lane);
      v16h b1 = load_b_frag(kt, 16, kk, CC, lane);
      s0 = WMMA_F16(a, b0, s0);
      s1 = WMMA_F16(a, b1, s1);
    }

    // ---- online softmax (per 16-lane half; row = r + 8*hl) --------------
    _Float16* pbuf = &lds_p[wave][0];
#pragma unroll
    for (int r = 0; r < 8; ++r) {
      float a0 = s0[r] * scale;
      float a1 = s1[r] * scale;
      float rm = fmaxf(a0, a1);
      for (int msk = 8; msk >= 1; msk >>= 1) rm = fmaxf(rm, __shfl_xor(rm, msk, 32));
      float mn = fmaxf(m[r], rm);
      float alpha = __expf(m[r] - mn);
      float p0 = __expf(a0 - mn);
      float p1 = __expf(a1 - mn);
      float rs = p0 + p1;
      for (int msk = 8; msk >= 1; msk >>= 1) rs += __shfl_xor(rs, msk, 32);
      l[r] = l[r] * alpha + rs;
      m[r] = mn;
#pragma unroll
      for (int t = 0; t < 8; ++t) o[t][r] *= alpha;
      int rowp = r + 8 * hl;
      pbuf[rowp * 32 + ln]      = (_Float16)p0;
      pbuf[rowp * 32 + 16 + ln] = (_Float16)p1;
    }

    // ---- O += P * V over this wave's 128-channel strip ------------------
    v16h ap = load_a_frag(pbuf, 0, 32, lane);
#pragma unroll
    for (int t = 0; t < 8; ++t) {
      v16h bv = load_b_frag(vb, colBase + t * 16, j0, NN, lane);
      o[t] = WMMA_F16(ap, bv, o[t]);
    }

    __syncthreads();     // everyone done reading lds_k[cur] before it is reused
  }

  // ---- normalize and write h [B,N,C] f16 --------------------------------
  _Float16* ho = hh + ((size_t)b * NN + row0) * CC;
#pragma unroll
  for (int r = 0; r < 8; ++r) {
    float inv = 1.0f / l[r];
    int rowp = r + 8 * hl;
#pragma unroll
    for (int t = 0; t < 8; ++t)
      ho[(size_t)rowp * CC + colBase + t * 16 + ln] = (_Float16)(o[t][r] * inv);
  }
}

// ---------------------------------------------------------------------------
// proj_out + residual: out[b,o,n] = x[b,o,n] + sum_c h[b,n,c]*wo[o,c] + bo[o].
// ---------------------------------------------------------------------------
__global__ __launch_bounds__(128) void proj_residual_kernel(const _Float16* __restrict__ hh,
                                                            const _Float16* __restrict__ woh,
                                                            const float* __restrict__ bo,
                                                            const float* __restrict__ x,
                                                            float* __restrict__ out) {
  const int b = blockIdx.z;
  const int colBase = blockIdx.x * 64;
  const int wave = threadIdx.x >> 5;
  const int lane = threadIdx.x & 31;
  const int hl = lane >> 4, ln = lane & 15;
  const int row0 = blockIdx.y * 64 + wave * 16;

  const _Float16* A = hh + ((size_t)b * NN + row0) * CC;

  v8f zero = {};
  v8f acc[4] = {zero, zero, zero, zero};
#pragma unroll
  for (int kk = 0; kk < CC; kk += 32) {
    v16h a = load_a_frag(A, kk, CC, lane);
#pragma unroll
    for (int t = 0; t < 4; ++t) {
      v16h bf = load_b_frag(woh, colBase + t * 16, kk, CC, lane);
      acc[t] = WMMA_F16(a, bf, acc[t]);
    }
  }

  const float* xb = x + (size_t)b * CC * NN;
  float* ob = out + (size_t)b * CC * NN;
#pragma unroll
  for (int t = 0; t < 4; ++t) {
    int col = colBase + t * 16 + ln;
    float bb = bo[col];
#pragma unroll
    for (int r = 0; r < 8; ++r) {
      size_t idx = (size_t)col * NN + row0 + r + 8 * hl;
      ob[idx] = xb[idx] + acc[t][r] + bb;
    }
  }
}

// ---------------------------------------------------------------------------
// Host-side orchestration.
// ---------------------------------------------------------------------------
extern "C" void kernel_launch(void* const* d_in, const int* in_sizes, int n_in,
                              void* d_out, int out_size, void* d_ws, size_t ws_size,
                              hipStream_t stream) {
  (void)in_sizes; (void)n_in; (void)out_size; (void)ws_size;

  const float* x      = (const float*)d_in[0];
  const float* norm_g = (const float*)d_in[1];
  const float* norm_b = (const float*)d_in[2];
  const float* wq     = (const float*)d_in[3];
  const float* bq     = (const float*)d_in[4];
  const float* wk     = (const float*)d_in[5];
  const float* bk     = (const float*)d_in[6];
  const float* wv     = (const float*)d_in[7];
  const float* bv     = (const float*)d_in[8];
  const float* wo     = (const float*)d_in[9];
  const float* bo     = (const float*)d_in[10];
  float* out = (float*)d_out;

  // Workspace layout (all 256B aligned).
  const size_t TOK = (size_t)BB * NN * CC;  // 8,388,608 f16 elements per tensor
  char* ws = (char*)d_ws;
  float*    gstats = (float*)ws;                               // 128*2 floats
  _Float16* t_h    = (_Float16*)(ws + 4096);
  _Float16* q_h    = t_h + TOK;
  _Float16* k_h    = q_h + TOK;
  _Float16* vT_h   = k_h + TOK;
  _Float16* h_h    = vT_h + TOK;
  _Float16* w_h    = h_h + TOK;                                // 4 * 512*512 f16

  // 1. GroupNorm statistics (one block per batch-group).
  gn_stats_kernel<<<dim3(BB * 32), dim3(256), 0, stream>>>(x, gstats);

  // 2. Normalize + transpose to token-major f16 [B,N,C].
  gn_apply_kernel<<<dim3((BB * CC * NN) / 256), dim3(256), 0, stream>>>(
      x, norm_g, norm_b, gstats, t_h);

  // 3. Weights -> f16.
  wconvert_kernel<<<dim3((4 * CC * CC) / 256), dim3(256), 0, stream>>>(
      wq, wk, wv, wo, w_h);

  // 4. Q, K, V projections (V stored transposed [B,C,N]).
  gemm_qkv_kernel<<<dim3(CC / 64, NN / 64, BB * 3), dim3(128), 0, stream>>>(
      t_h, w_h, bq, bk, bv, q_h, k_h, vT_h);

  // 5. Flash attention (async double-buffered K staging).
  flash_attn_kernel<<<dim3(NN / 16, BB), dim3(128), 0, stream>>>(
      q_h, k_h, vT_h, h_h);

  // 6. Output projection + residual, transposed back to [B,C,H,W].
  proj_residual_kernel<<<dim3(CC / 64, NN / 64, BB), dim3(128), 0, stream>>>(
      h_h, w_h + (size_t)3 * CC * CC, bo, x, out);
}
